// MoChA_43662637532057
// MI455X (gfx1250) — matrix operand order repaired and checked by reading, not compiled, not run on hardware
//
#include <hip/hip_runtime.h>
#include <hip/hip_bf16.h>

typedef __attribute__((ext_vector_type(16))) _Float16 v16h;
typedef __attribute__((ext_vector_type(8)))  float    v8f;

#define B_ 16
#define S_ 4096
#define D_ 1024
#define NEGINF (-1e9f)

// ---------------------------------------------------------------------------
// Kernel 0: wn = v_g * v_v / ||v_v||   (single block)
// ---------------------------------------------------------------------------
__global__ void k_wnorm(const float* __restrict__ vv, const float* __restrict__ vg,
                        float* __restrict__ wn) {
    __shared__ float red[256];
    float acc = 0.f;
    for (int d = threadIdx.x; d < D_; d += 256) { float x = vv[d]; acc += x * x; }
    red[threadIdx.x] = acc;
    __syncthreads();
    for (int off = 128; off > 0; off >>= 1) {
        if (threadIdx.x < off) red[threadIdx.x] += red[threadIdx.x + off];
        __syncthreads();
    }
    float scale = vg[0] * rsqrtf(red[0]);
    for (int d = threadIdx.x; d < D_; d += 256) wn[d] = vv[d] * scale;
}

// ---------------------------------------------------------------------------
// Kernel 1: energies. One wave per (b,s); both 1024-dots fused so k[b,0,s,:]
// and k[b,1,s,:] are each streamed exactly once (512 MB total). Each lane
// loads float4 -> global_load_b128, 512B per wave per issue on the hot stream.
// grid = B * S/8 blocks of 256 threads (8 waves). q+bias / weights in LDS.
// ---------------------------------------------------------------------------
__global__ void k_energy(const float* __restrict__ q, const float* __restrict__ k,
                         const float* __restrict__ b_mono, const float* __restrict__ b_chunk,
                         const float* __restrict__ wn, const float* __restrict__ w_chunk,
                         float* __restrict__ e_mono, float* __restrict__ e_chunk) {
    __shared__ float qm[D_], qc[D_], wm[D_], wc[D_];
    int b = blockIdx.x >> 9;              // 512 blocks per batch
    int s_base = (blockIdx.x & 511) * 8;
    // stage q+bias and weight vectors (float4 loads)
    {
        const float4* q0 = (const float4*)(q + (b * 2 + 0) * D_);
        const float4* q1 = (const float4*)(q + (b * 2 + 1) * D_);
        const float4* bm = (const float4*)b_mono;
        const float4* bc = (const float4*)b_chunk;
        const float4* w0 = (const float4*)wn;
        const float4* w1 = (const float4*)w_chunk;
        for (int i = threadIdx.x; i < D_ / 4; i += 256) {
            float4 a = q0[i], bb = bm[i];
            ((float4*)qm)[i] = make_float4(a.x + bb.x, a.y + bb.y, a.z + bb.z, a.w + bb.w);
            a = q1[i]; bb = bc[i];
            ((float4*)qc)[i] = make_float4(a.x + bb.x, a.y + bb.y, a.z + bb.z, a.w + bb.w);
            ((float4*)wm)[i] = w0[i];
            ((float4*)wc)[i] = w1[i];
        }
    }
    __syncthreads();
    int wave = threadIdx.x >> 5, lane = threadIdx.x & 31;
    int s = s_base + wave;
    const float4* k0 = (const float4*)(k + (((size_t)b * 2 + 0) * S_ + s) * D_);
    const float4* k1 = (const float4*)(k + (((size_t)b * 2 + 1) * S_ + s) * D_);
    float am = 0.f, ac = 0.f;
    for (int i = 0; i < D_ / 4; i += 32) {
        int idx = i + lane;               // coalesced 512B (b128 per lane)
        int d = idx * 4;
        float4 kv = k0[idx];
        am += tanhf(qm[d + 0] + kv.x) * wm[d + 0];
        am += tanhf(qm[d + 1] + kv.y) * wm[d + 1];
        am += tanhf(qm[d + 2] + kv.z) * wm[d + 2];
        am += tanhf(qm[d + 3] + kv.w) * wm[d + 3];
        kv = k1[idx];
        ac += tanhf(qc[d + 0] + kv.x) * wc[d + 0];
        ac += tanhf(qc[d + 1] + kv.y) * wc[d + 1];
        ac += tanhf(qc[d + 2] + kv.z) * wc[d + 2];
        ac += tanhf(qc[d + 3] + kv.w) * wc[d + 3];
    }
    // wave32 butterfly reduction
    for (int off = 16; off > 0; off >>= 1) {
        am += __shfl_xor(am, off, 32);
        ac += __shfl_xor(ac, off, 32);
    }
    if (lane == 0) { e_mono[b * S_ + s] = am; e_chunk[b * S_ + s] = ac; }
}

// ---------------------------------------------------------------------------
// Kernel 2: per-batch sequential part. One block (256 thr) per batch.
//  phase1: p = sigmoid(e+bias+noise) w/ mask & end-mask; masked ec  -> LDS
//  phase2: wave-0 inclusive scan of a <- shifted*a + prev_att via
//          linear-function composition (affine scan, 5 shfl steps/chunk)
//  phase3: trailing moving max (w=8) + exp                         -> LDS
//  phase4: trailing moving sum (denominator), ratio = alpha/denom  -> LDS
//  phase5: forward moving sum of ratio, beta -> d_out beta slot
// LDS: 4 * 16KB aliased buffers = 64 KB.
// ---------------------------------------------------------------------------
__global__ void k_scan_beta(const float* __restrict__ e_mono, const float* __restrict__ e_chunk,
                            const float* __restrict__ noise, const float* __restrict__ prev_att,
                            const int* __restrict__ k_len, const float* __restrict__ v_b,
                            const float* __restrict__ r0, float* __restrict__ beta_out) {
    __shared__ float sm[4 * S_];
    float* p_arr     = sm;            // dead after phase2
    float* ec_arr    = sm + S_;
    float* alpha_arr = sm + 2 * S_;
    float* expe_arr  = sm + 3 * S_;
    float* ratio_arr = sm;            // reuses p_arr storage (after barrier)

    int b  = blockIdx.x;
    int kl = k_len[b];
    float bias = v_b[0] + r0[0];

    // phase 1
    for (int s = threadIdx.x; s < S_; s += blockDim.x) {
        bool msk = s < kl;
        float e = msk ? (e_mono[b * S_ + s] + bias) : NEGINF;
        float x = e + noise[b * S_ + s];
        float p = (s == kl - 1) ? 1.0f : 1.0f / (1.0f + __expf(-x));
        p_arr[s]  = p;
        ec_arr[s] = msk ? e_chunk[b * S_ + s] : NEGINF;
    }
    __syncthreads();

    // phase 2: affine scan on wave 0
    if (threadIdx.x < 32) {
        int lane = threadIdx.x;
        float carry = 0.f;
        for (int c = 0; c < S_; c += 32) {
            int s = c + lane;
            float p_s   = p_arr[s];
            float mcoef = (s == 0) ? 1.0f : (1.0f - p_arr[s - 1]);
            float cadd  = prev_att[b * S_ + s];
            for (int off = 1; off < 32; off <<= 1) {       // inclusive scan of (m,c)
                float mo = __shfl_up(mcoef, off, 32);
                float co = __shfl_up(cadd,  off, 32);
                if (lane >= off) { cadd = mcoef * co + cadd; mcoef = mcoef * mo; }
            }
            float a_s = mcoef * carry + cadd;
            alpha_arr[s] = p_s * a_s;
            carry = __shfl(a_s, 31, 32);
        }
    }
    __syncthreads();

    // phase 3: m = movmax(ec, 8);  exp_e = mask ? exp(ec - m) : 0
    for (int s = threadIdx.x; s < S_; s += blockDim.x) {
        float mx = NEGINF;
        int j0 = (s - 7 < 0) ? 0 : s - 7;
        for (int j = j0; j <= s; ++j) mx = fmaxf(mx, ec_arr[j]);
        expe_arr[s] = (s < kl) ? __expf(ec_arr[s] - mx) : 0.0f;
    }
    __syncthreads();

    // phase 4: denom = max(movsum(exp_e,7,0), 1e-20); ratio = alpha/denom
    for (int s = threadIdx.x; s < S_; s += blockDim.x) {
        float den = 0.f;
        int j0 = (s - 7 < 0) ? 0 : s - 7;
        for (int j = j0; j <= s; ++j) den += expe_arr[j];
        den = fmaxf(den, 1e-20f);
        ratio_arr[s] = alpha_arr[s] / den;
    }
    __syncthreads();

    // phase 5: beta = mask ? exp_e * movsum(ratio, 0, 7) : 0
    for (int s = threadIdx.x; s < S_; s += blockDim.x) {
        float sum = 0.f;
        int j1 = (s + 7 > S_ - 1) ? S_ - 1 : s + 7;
        for (int j = s; j <= j1; ++j) sum += ratio_arr[j];
        beta_out[b * S_ + s] = (s < kl) ? expe_arr[s] * sum : 0.0f;
    }
}

// ---------------------------------------------------------------------------
// Kernel 3: context[b,d] = sum_s beta[b,s] * v[b,s,d]  via WMMA f16->f32.
// One wave per (b, 16-wide d-tile): A = v-tile (M=d, K=s), B = beta chunk
// replicated across all N columns. TWO independent f32 accumulators (64 s per
// iteration) break the WMMA C->C RAW chain so the matrix ops overlap with the
// next chunk's loads; partials summed exactly in f32 at the end.
// A layout (ISA 16-bit A 16x32): lane<16 row M=lane, K in {0..7,16..23};
// lane>=16 same M, K in {8..15,24..31}. B layout (32x16): element hh ->
// K = 8*(lane>>4) + (hh&7) (+16 for hh>=8), N = lane&15. D: lane0 col N=0 M=0..7,
// lane16 col N=0 M=8..15.
// ---------------------------------------------------------------------------
__device__ __forceinline__ void load_frags(const float* __restrict__ vb,
                                           const float* __restrict__ betab,
                                           int s0, int grp, v16h& a, v16h& bb) {
#pragma unroll
    for (int hh = 0; hh < 16; ++hh) {
        int j = hh >> 1, h = hh & 1;
        int KA = ((j & 4) ? 16 : 0) + 8 * grp + 2 * (j & 3) + h;   // A: row s0+KA
        a[hh]  = (_Float16)vb[(size_t)(s0 + KA) * D_];
        int KB = ((hh & 8) ? 16 : 0) + 8 * grp + (hh & 7);         // B: beta[s0+KB]
        bb[hh] = (_Float16)betab[s0 + KB];
    }
}

__global__ void k_context(const float* __restrict__ v, const float* __restrict__ beta,
                          float* __restrict__ ctx) {
    int b    = blockIdx.x >> 6;       // 64 d-tiles of 16
    int d0   = (blockIdx.x & 63) << 4;
    int lane = threadIdx.x;
    int grp  = lane >> 4;
    int m    = lane & 15;
    const float* vb    = v + ((size_t)b * S_) * D_ + d0 + m;
    const float* betab = beta + b * S_;
    v8f c0 = {}, c1 = {};
    for (int s0 = 0; s0 < S_; s0 += 64) {
        v16h a0, b0, a1, b1;
        load_frags(vb, betab, s0,      grp, a0, b0);
        load_frags(vb, betab, s0 + 32, grp, a1, b1);
        c0 = __builtin_amdgcn_wmma_f32_16x16x32_f16(false, a0, false, b0,
                                                    (short)0, c0, false, false);
        c1 = __builtin_amdgcn_wmma_f32_16x16x32_f16(false, a1, false, b1,
                                                    (short)0, c1, false, false);
    }
    v8f c;
#pragma unroll
    for (int rr = 0; rr < 8; ++rr) c[rr] = c0[rr] + c1[rr];
    // D tile column N=0: lane 0 holds M=0..7, lane 16 holds M=8..15
    if ((lane & 15) == 0) {
        int mbase = (lane >> 4) * 8;
#pragma unroll
        for (int rr = 0; rr < 8; ++rr) ctx[b * D_ + d0 + mbase + rr] = c[rr];
    }
}

// ---------------------------------------------------------------------------
extern "C" void kernel_launch(void* const* d_in, const int* in_sizes, int n_in,
                              void* d_out, int out_size, void* d_ws, size_t ws_size,
                              hipStream_t stream) {
    const float* q        = (const float*)d_in[0];   // [16,2,1024]
    const float* k        = (const float*)d_in[1];   // [16,2,4096,1024]
    const float* v        = (const float*)d_in[2];   // [16,4096,1024]
    const float* prev_att = (const float*)d_in[3];   // [16,4096]
    const float* noise    = (const float*)d_in[4];   // [16,4096]
    const float* b_mono   = (const float*)d_in[5];   // [1024]
    const float* v_v      = (const float*)d_in[6];   // [1024]
    const float* v_g      = (const float*)d_in[7];   // [1]
    const float* v_b      = (const float*)d_in[8];   // [1]
    const float* r0       = (const float*)d_in[9];   // [1]
    const float* b_chunk  = (const float*)d_in[10];  // [1024]
    const float* w_chunk  = (const float*)d_in[11];  // [1024]
    const int*   k_len    = (const int*)d_in[12];    // [16]

    float* out      = (float*)d_out;
    float* ctx_out  = out;                 // [16,1024]
    float* beta_out = out + B_ * D_;       // [16,1,4096]

    float* wn      = (float*)d_ws;         // [1024]
    float* e_mono  = wn + D_;              // [16*4096]
    float* e_chunk = e_mono + B_ * S_;     // [16*4096]

    k_wnorm<<<1, 256, 0, stream>>>(v_v, v_g, wn);
    k_energy<<<B_ * (S_ / 8), 256, 0, stream>>>(q, k, b_mono, b_chunk, wn, w_chunk,
                                                e_mono, e_chunk);
    k_scan_beta<<<B_, 256, 0, stream>>>(e_mono, e_chunk, noise, prev_att, k_len,
                                        v_b, r0, beta_out);
    k_context<<<B_ * (D_ / 16), 32, 0, stream>>>(v, beta_out, ctx_out);
}
